// LMHSA_49409303773798
// MI455X (gfx1250) — matrix operand order repaired and checked by reading, not compiled, not run on hardware
//
#include <hip/hip_runtime.h>
#include <hip/hip_bf16.h>

typedef __attribute__((ext_vector_type(16))) _Float16 v16h;
typedef __attribute__((ext_vector_type(8)))  float    v8f;

namespace lmhsa {

constexpr int BATCH = 16;
constexpr int CCH   = 512;
constexpr int NSP   = 56*56;     // 3136
constexpr int NKD   = 49;
constexpr int HEADS = 8;
constexpr int HID   = 24;
constexpr float SCALEF = 0.125f; // 64^-0.5
constexpr long long SSP = (long long)NSP*NKD; // 153664
constexpr int CHUNK  = 8192;
constexpr int NCHUNK = 19;       // ceil(153664/8192)

union AFrag { v16h v; unsigned u[8]; };

// ---------------------------------------------------------------------------
// Generic batched WMMA GEMM.  C[M x Ntot] = A[M x K] * B (N-major: Ntot rows
// of K halves).  K must be a multiple of 32.  Batch z -> (b = z/batchDiv,
// h = z%batchDiv), operand offsets b*s?b + h*s?h (element units).
// ---------------------------------------------------------------------------
template<bool OUTF16>
__global__ __launch_bounds__(256)
void k_gemm_wmma(const _Float16* __restrict__ A, const _Float16* __restrict__ Bm,
                 void* __restrict__ Cm, const float* __restrict__ bias,
                 int M, int Ntot, int K, int lda, int ldb, int ldc,
                 int batchDiv,
                 long long sAb, long long sAh, long long sBb, long long sBh,
                 long long sCb, long long sCh)
{
  __shared__ unsigned As[128*17];
  __shared__ unsigned Bs[64*17];

  const int z  = blockIdx.z;
  const int bb = z / batchDiv, hh = z % batchDiv;
  const _Float16* Ab = A  + bb*sAb + hh*sAh;
  const _Float16* Bb = Bm + bb*sBb + hh*sBh;
  const long long coff = bb*sCb + hh*sCh;

  const int m0 = blockIdx.x * 128;
  const int n0 = blockIdx.y * 64;

  const int tid  = threadIdx.x;
  const int lane = tid & 31;
  const int wid  = tid >> 5;
  const int wm   = wid & 3;   // 4 waves along M
  const int wn   = wid >> 2;  // 2 waves along N

  v8f acc[2][2] = {};

  const int la_m = tid >> 1, la_p = tid & 1;   // A loader: row, 16-half seg
  const int lb_n = tid >> 2, lb_q = tid & 3;   // B loader: n-row, uint4 seg

  const int lmod = lane & 15;
  const int lhi  = lane >> 4;
  const int kb8  = lhi * 4;   // A frag k-pair base
  const int kb16 = lhi * 8;   // unused; B frag also uses kb8

  for (int k0 = 0; k0 < K; k0 += 32) {
    { // stage A tile (128 x 32 halves)
      const int gr = m0 + la_m;
      uint4 q0 = make_uint4(0,0,0,0), q1 = make_uint4(0,0,0,0);
      if (gr < M) {
        const uint4* p = reinterpret_cast<const uint4*>(Ab + (long long)gr*lda + k0 + la_p*16);
        q0 = p[0]; q1 = p[1];
        if (k0 + 32 < K)
          __builtin_prefetch(Ab + (long long)gr*lda + k0 + 32 + la_p*16, 0, 1);
      }
      unsigned* d = &As[la_m*17 + la_p*8];
      d[0]=q0.x; d[1]=q0.y; d[2]=q0.z; d[3]=q0.w;
      d[4]=q1.x; d[5]=q1.y; d[6]=q1.z; d[7]=q1.w;
    }
    { // stage B tile (64 n-rows x 32 halves)
      const int gn = n0 + lb_n;
      uint4 q = make_uint4(0,0,0,0);
      if (gn < Ntot) {
        const uint4* p = reinterpret_cast<const uint4*>(Bb + (long long)gn*ldb + k0 + lb_q*8);
        q = p[0];
      }
      unsigned* d = &Bs[lb_n*17 + lb_q*4];
      d[0]=q.x; d[1]=q.y; d[2]=q.z; d[3]=q.w;
    }
    __syncthreads();

    AFrag a[2], b[2];
    #pragma unroll
    for (int i = 0; i < 2; ++i) {
      const unsigned* pa = &As[(wm*32 + i*16 + lmod)*17];
      #pragma unroll
      for (int j = 0; j < 8; ++j)
        a[i].u[j] = pa[((j & 4) ? 8 : 0) + kb8 + (j & 3)];
    }
    #pragma unroll
    for (int i = 0; i < 2; ++i) {
      const unsigned* pb = &Bs[(wn*32 + i*16 + lmod)*17];
      #pragma unroll
      for (int j = 0; j < 8; ++j)
        b[i].u[j] = pb[((j & 4) ? 8 : 0) + kb8 + (j & 3)];
    }
    #pragma unroll
    for (int i = 0; i < 2; ++i)
      #pragma unroll
      for (int jn = 0; jn < 2; ++jn)
        acc[i][jn] = __builtin_amdgcn_wmma_f32_16x16x32_f16(
            false, a[i].v, false, b[jn].v, (short)0, acc[i][jn], false, false);

    __syncthreads();
  }

  #pragma unroll
  for (int i = 0; i < 2; ++i) {
    #pragma unroll
    for (int jn = 0; jn < 2; ++jn) {
      const int col = n0 + wn*32 + jn*16 + lmod;
      if (col >= Ntot) continue;
      #pragma unroll
      for (int e = 0; e < 8; ++e) {
        const int row = m0 + wm*32 + i*16 + lhi*8 + e;
        if (row >= M) continue;
        float v = acc[i][jn][e];
        if (bias) v += bias[col];
        const long long off = coff + (long long)row*ldc + col;
        if (OUTF16) ((_Float16*)Cm)[off] = (_Float16)v;
        else        ((float*)Cm)[off]    = v;
      }
    }
  }
}

// ---------------- x (B,C,N) -> xt (B*N, C) f16 -----------------------------
__global__ __launch_bounds__(256) void k_transpose_x(const float* __restrict__ x,
                                                     _Float16* __restrict__ xt)
{
  __shared__ float tile[32][33];
  const int b = blockIdx.z;
  const int n0 = blockIdx.x*32, c0 = blockIdx.y*32;
  const int tx = threadIdx.x, ty = threadIdx.y;
  #pragma unroll
  for (int i=0;i<4;i++)
    tile[ty+i*8][tx] = x[((long long)b*CCH + (c0+ty+i*8))*NSP + n0 + tx];
  __syncthreads();
  #pragma unroll
  for (int i=0;i<4;i++)
    xt[((long long)b*NSP + (n0+ty+i*8))*CCH + c0 + tx] = (_Float16)tile[tx][ty+i*8];
}

// ---------------- depthwise 8x8/8 downsample -> kvx (B*49, C) f16 ----------
__global__ __launch_bounds__(256) void k_downsample(const float* __restrict__ x,
                                                    const float* __restrict__ dw,
                                                    _Float16* __restrict__ kvx)
{
  const int gid = blockIdx.x*256 + threadIdx.x;
  if (gid >= BATCH*NKD*CCH) return;
  const int c = gid % CCH;
  const int m = (gid / CCH) % NKD;
  const int b = gid / (CCH*NKD);
  const int my = m / 7, mx = m % 7;
  const float* xp = x + ((long long)b*CCH + c)*NSP;
  const float* wp = dw + (long long)c*64;
  float acc = 0.f;
  #pragma unroll
  for (int i = 0; i < 8; ++i)
    #pragma unroll
    for (int j = 0; j < 8; ++j)
      acc += xp[(my*8+i)*56 + mx*8 + j] * wp[i*8+j];
  kvx[((long long)b*NKD + m)*CCH + c] = (_Float16)acc;
}

// ---------------- weight (K,Nt) row-major -> f16 N-major (Nt rows of K) ----
__global__ void k_convert_wt(const float* __restrict__ w, _Float16* __restrict__ wt,
                             int K, int Nt)
{
  const int idx = blockIdx.x*256 + threadIdx.x;
  if (idx >= K*Nt) return;
  const int k = idx % K, n = idx / K;
  wt[idx] = (_Float16)w[(long long)k*Nt + n];
}

// ------------- split kv (B,49,1024) -> k_h [z][m][d], v_h [z][d][m], pad 64 -
__global__ void k_build_kv(const float* __restrict__ kv,
                           _Float16* __restrict__ kh, _Float16* __restrict__ vh)
{
  const int idx = blockIdx.x*256 + threadIdx.x;
  if (idx >= 128*4096) return;
  const int d = idx & 63, m = (idx>>6)&63, z = idx>>12;
  const int b = z>>3, h = z&7;
  _Float16 kvv = (_Float16)0.f, vvv = (_Float16)0.f;
  if (m < NKD) {
    const float* row = kv + ((long long)b*NKD + m)*1024 + h*64 + d;
    kvv = (_Float16)row[0];
    vvv = (_Float16)row[512];
  }
  kh[(long long)z*4096 + m*64 + d] = kvv;
  vh[(long long)z*4096 + d*64 + m] = vvv;
}

// ---------------- softmax over 49 with scale + rel_bias --------------------
__global__ __launch_bounds__(256) void k_softmax(const float* __restrict__ logits,
                                                 const float* __restrict__ rel_bias,
                                                 _Float16* __restrict__ attn)
{
  const long long row = ((long long)blockIdx.x*256 + threadIdx.x) >> 5;
  const int lane = threadIdx.x & 31;
  if (row >= (long long)BATCH*HEADS*NSP) return;
  const int n = (int)(row % NSP);
  const float* lp = logits + row*NKD;
  const float* rb = rel_bias + (long long)n*NKD;
  float v0 = -1e30f, v1 = -1e30f;
  if (lane < NKD)      v0 = lp[lane]   *SCALEF + rb[lane];
  if (lane+32 < NKD)   v1 = lp[lane+32]*SCALEF + rb[lane+32];
  float mx = fmaxf(v0, v1);
  #pragma unroll
  for (int s=16; s>0; s>>=1) mx = fmaxf(mx, __shfl_xor(mx, s, 32));
  float e0 = (lane<NKD)    ? __expf(v0-mx) : 0.f;
  float e1 = (lane+32<NKD) ? __expf(v1-mx) : 0.f;
  float sm = e0+e1;
  #pragma unroll
  for (int s=16; s>0; s>>=1) sm += __shfl_xor(sm, s, 32);
  const float inv = 1.f/sm;
  _Float16* op = attn + row*NKD;
  if (lane < NKD)    op[lane]    = (_Float16)(e0*inv);
  if (lane+32 < NKD) op[lane+32] = (_Float16)(e1*inv);
}

// ---------------- block partial-sum helper ---------------------------------
__device__ __forceinline__ void block_partials(float ls, float lq, float* psum,
                                               float* psq, long long slot)
{
  #pragma unroll
  for (int s=16; s>0; s>>=1){ ls += __shfl_xor(ls,s,32); lq += __shfl_xor(lq,s,32); }
  __shared__ float ssum[8], ssq[8];
  const int lane = threadIdx.x & 31, w = threadIdx.x >> 5;
  if (lane==0){ ssum[w]=ls; ssq[w]=lq; }
  __syncthreads();
  if (threadIdx.x==0){
    float a=0.f,b=0.f;
    for (int i=0;i<8;i++){ a+=ssum[i]; b+=ssq[i]; }
    psum[slot]=a; psq[slot]=b;
  }
}

// ---------------- expand 1x1 conv (8->24) + partial stats ------------------
__global__ __launch_bounds__(256) void k_expand(const _Float16* __restrict__ attn,
                                                const float* __restrict__ w,
                                                _Float16* __restrict__ a1,
                                                float* __restrict__ psum,
                                                float* __restrict__ psq)
{
  const int z = blockIdx.y;          // b*24+cc
  const int b = z/24, cc = z%24;
  float wl[8];
  #pragma unroll
  for (int h=0;h<8;h++) wl[h] = w[cc*8+h];
  const _Float16* ap = attn + (long long)b*8*SSP;
  _Float16* op = a1 + (long long)z*SSP;
  float ls=0.f, lq=0.f;
  const int base = blockIdx.x*CHUNK;
  for (int i = threadIdx.x; i < CHUNK; i += 256) {
    const int s = base + i;
    if (s >= SSP) break;
    float v = 0.f;
    #pragma unroll
    for (int h=0;h<8;h++) v += wl[h]*(float)ap[(long long)h*SSP + s];
    op[s] = (_Float16)v;
    ls += v; lq += v*v;
  }
  block_partials(ls, lq, psum, psq, (long long)z*NCHUNK + blockIdx.x);
}

// ---------------- depthwise 3x3 (SAME) over (3136,49) + partial stats ------
__global__ __launch_bounds__(256) void k_dwconv(const _Float16* __restrict__ a1,
                                                const float* __restrict__ w,
                                                _Float16* __restrict__ a2,
                                                float* __restrict__ psum,
                                                float* __restrict__ psq)
{
  const int z = blockIdx.y;
  const int cc = z % 24;
  float wl[9];
  #pragma unroll
  for (int i=0;i<9;i++) wl[i]=w[cc*9+i];
  const _Float16* ip = a1 + (long long)z*SSP;
  _Float16* op = a2 + (long long)z*SSP;
  float ls=0.f, lq=0.f;
  const int base = blockIdx.x*CHUNK;
  for (int i = threadIdx.x; i < CHUNK; i += 256) {
    const int s = base + i;
    if (s >= SSP) break;
    const int n = s / NKD, m = s % NKD;
    float v = 0.f;
    #pragma unroll
    for (int dn=0; dn<3; dn++){
      const int nn = n+dn-1;
      if (nn < 0 || nn >= NSP) continue;
      #pragma unroll
      for (int dm=0; dm<3; dm++){
        const int mm = m+dm-1;
        if (mm < 0 || mm >= NKD) continue;
        v += wl[dn*3+dm]*(float)ip[(long long)nn*NKD + mm];
      }
    }
    op[s] = (_Float16)v;
    ls += v; lq += v*v;
  }
  block_partials(ls, lq, psum, psq, (long long)z*NCHUNK + blockIdx.x);
}

// ---------------- channel reduce 1x1 (24->8) + partial stats ---------------
__global__ __launch_bounds__(256) void k_chreduce(const _Float16* __restrict__ a2,
                                                  const float* __restrict__ w,
                                                  _Float16* __restrict__ a3,
                                                  float* __restrict__ psum,
                                                  float* __restrict__ psq)
{
  const int z = blockIdx.y;           // b*8+hh
  const int b = z/8, hh = z%8;
  float wl[24];
  #pragma unroll
  for (int i=0;i<24;i++) wl[i]=w[hh*24+i];
  const _Float16* ip = a2 + (long long)b*24*SSP;
  _Float16* op = a3 + (long long)z*SSP;
  float ls=0.f, lq=0.f;
  const int base = blockIdx.x*CHUNK;
  for (int i = threadIdx.x; i < CHUNK; i += 256) {
    const int s = base + i;
    if (s >= SSP) break;
    float v = 0.f;
    #pragma unroll
    for (int cc=0; cc<24; cc++) v += wl[cc]*(float)ip[(long long)cc*SSP + s];
    op[s] = (_Float16)v;
    ls += v; lq += v*v;
  }
  block_partials(ls, lq, psum, psq, (long long)z*NCHUNK + blockIdx.x);
}

// ---------------- reduce partials -> (mean, inv_std) per (b, group) --------
__global__ __launch_bounds__(128) void k_gn_stats(const float* __restrict__ psum,
                                                  const float* __restrict__ psq,
                                                  int nch, int gs, int ngroups,
                                                  float count, float* __restrict__ stats)
{
  const int bid = blockIdx.x;               // b*ngroups + g
  const int b = bid / ngroups, g = bid % ngroups;
  const int total = gs * NCHUNK;
  float ls=0.f, lq=0.f;
  for (int i = threadIdx.x; i < total; i += 128) {
    const int ch = g*gs + i/NCHUNK;
    const long long idx = ((long long)b*nch + ch)*NCHUNK + (i % NCHUNK);
    ls += psum[idx]; lq += psq[idx];
  }
  #pragma unroll
  for (int s=16; s>0; s>>=1){ ls += __shfl_xor(ls,s,32); lq += __shfl_xor(lq,s,32); }
  __shared__ float ssum[4], ssq[4];
  const int lane = threadIdx.x & 31, w = threadIdx.x >> 5;
  if (lane==0){ ssum[w]=ls; ssq[w]=lq; }
  __syncthreads();
  if (threadIdx.x==0){
    float a=0.f, q=0.f;
    for (int i=0;i<4;i++){ a+=ssum[i]; q+=ssq[i]; }
    const float mean = a / count;
    const float var  = q / count - mean*mean;
    stats[bid*2]   = mean;
    stats[bid*2+1] = rsqrtf(var + 1e-5f);
  }
}

// ---------------- apply GN (+optional swish) in place ----------------------
__global__ __launch_bounds__(256) void k_gn_apply(_Float16* __restrict__ data,
                                                  const float* __restrict__ stats,
                                                  const float* __restrict__ scale,
                                                  const float* __restrict__ bias,
                                                  int nch, int gs, int swish,
                                                  long long total)
{
  const long long stride = (long long)gridDim.x*256;
  for (long long idx = (long long)blockIdx.x*256 + threadIdx.x; idx < total; idx += stride) {
    const long long chs = idx / SSP;       // b*nch + ch
    const int ch = (int)(chs % nch);
    const int b  = (int)(chs / nch);
    const int sid = (b*(nch/gs) + ch/gs)*2;
    const float mean = stats[sid], inv = stats[sid+1];
    float v = ((float)data[idx] - mean)*inv*scale[ch] + bias[ch];
    if (swish) v = v / (1.f + __expf(-v));
    data[idx] = (_Float16)v;
  }
}

// ------------- GN3 apply + pad NK 49->64, f16 out for attn@v GEMM ----------
__global__ __launch_bounds__(256) void k_gn3_pad(const _Float16* __restrict__ a3,
                                                 const float* __restrict__ stats,
                                                 const float* __restrict__ sc,
                                                 const float* __restrict__ bi,
                                                 _Float16* __restrict__ outp)
{
  const long long idx = (long long)blockIdx.x*256 + threadIdx.x;
  if (idx >= (long long)128*NSP*64) return;
  const int m = (int)(idx & 63);
  const long long r = idx >> 6;            // (b*8+hh)*NSP + n
  const int n  = (int)(r % NSP);
  const int zh = (int)(r / NSP);
  const int hh = zh & 7, b = zh >> 3;
  _Float16 v = (_Float16)0.f;
  if (m < NKD) {
    const float mean = stats[b*2], inv = stats[b*2+1];
    const float x = ((float)a3[(long long)zh*SSP + (long long)n*NKD + m] - mean)*inv*sc[hh] + bi[hh];
    v = (_Float16)x;
  }
  outp[idx] = v;
}

} // namespace lmhsa

extern "C" void kernel_launch(void* const* d_in, const int* in_sizes, int n_in,
                              void* d_out, int out_size, void* d_ws, size_t ws_size,
                              hipStream_t stream) {
  using namespace lmhsa;
  const float* x        = (const float*)d_in[0];
  const float* q_w      = (const float*)d_in[1];
  const float* down_w   = (const float*)d_in[2];
  const float* kv_w     = (const float*)d_in[3];
  const float* proj_w   = (const float*)d_in[4];
  const float* proj_b   = (const float*)d_in[5];
  const float* rel_bias = (const float*)d_in[6];
  const float* expand_w = (const float*)d_in[7];
  const float* gn1_s    = (const float*)d_in[8];
  const float* gn1_b    = (const float*)d_in[9];
  const float* dw_w     = (const float*)d_in[10];
  const float* gn2_s    = (const float*)d_in[11];
  const float* gn2_b    = (const float*)d_in[12];
  const float* reduce_w = (const float*)d_in[13];
  const float* gn3_s    = (const float*)d_in[14];
  const float* gn3_b    = (const float*)d_in[15];
  float* out = (float*)d_out;

  const long long BN = (long long)BATCH*NSP;   // 50176
  size_t off = 0;
  auto alloc = [&](size_t bytes)->char* {
    char* p = (char*)d_ws + off;
    off += (bytes + 255) & ~(size_t)255;
    return p;
  };
  _Float16* xt    = (_Float16*)alloc(BN*512*2);
  _Float16* qh    = (_Float16*)alloc(BN*512*2);
  _Float16* kvxh  = (_Float16*)alloc(784ULL*512*2);
  float*    kvf   = (float*)  alloc(784ULL*1024*4);
  _Float16* wqt   = (_Float16*)alloc(512ULL*512*2);
  _Float16* wkvt  = (_Float16*)alloc(512ULL*1024*2);
  _Float16* wpt   = (_Float16*)alloc(512ULL*512*2);
  _Float16* kh    = (_Float16*)alloc(128ULL*4096*2);
  _Float16* vh    = (_Float16*)alloc(128ULL*4096*2);
  float*    logit = (float*)  alloc(128ULL*NSP*NKD*4);
  _Float16* attnp = (_Float16*)alloc(128ULL*NSP*NKD*2);
  _Float16* a1    = (_Float16*)alloc(16ULL*24*SSP*2);
  _Float16* a2    = (_Float16*)alloc(16ULL*24*SSP*2);
  _Float16* a3    = (_Float16*)alloc(16ULL*8*SSP*2);
  _Float16* avh   = (_Float16*)alloc(128ULL*NSP*64*2);
  _Float16* aoh   = (_Float16*)alloc(BN*512*2);
  float*    psum  = (float*)  alloc(16ULL*24*NCHUNK*4);
  float*    psq   = (float*)  alloc(16ULL*24*NCHUNK*4);
  float*    stats = (float*)  alloc(256*4);
  if (off > ws_size) return;   // scratch too small: bail safely

  // 1) x (B,C,N) -> xt (B*N, C) f16
  k_transpose_x<<<dim3(NSP/32, 512/32, BATCH), dim3(32,8), 0, stream>>>(x, xt);
  // 2) depthwise 8x8 stride-8 downsample -> kvx f16
  k_downsample<<<(BATCH*NKD*512 + 255)/256, 256, 0, stream>>>(x, down_w, kvxh);
  // 3) weights -> f16 N-major
  k_convert_wt<<<(512*512 + 255)/256, 256, 0, stream>>>(q_w,    wqt,  512, 512);
  k_convert_wt<<<(512*1024 + 255)/256, 256, 0, stream>>>(kv_w,  wkvt, 512, 1024);
  k_convert_wt<<<(512*512 + 255)/256, 256, 0, stream>>>(proj_w, wpt,  512, 512);
  // 4) kv GEMM: (784 x 512) * (512 x 1024) -> f32
  k_gemm_wmma<false><<<dim3(7,16,1), 256, 0, stream>>>(
      kvxh, wkvt, kvf, nullptr, 784, 1024, 512, 512, 512, 1024,
      1, 0,0, 0,0, 0,0);
  // 5) split/pad k,v per head
  k_build_kv<<<(128*4096)/256, 256, 0, stream>>>(kvf, kh, vh);
  // 6) q GEMM: (50176 x 512) * (512 x 512) -> f16
  k_gemm_wmma<true><<<dim3(392,8,1), 256, 0, stream>>>(
      xt, wqt, qh, nullptr, (int)BN, 512, 512, 512, 512, 512,
      1, 0,0, 0,0, 0,0);
  // 7) logits: per (b,h): (3136 x 64) * k^T -> (3136 x 49) f32
  k_gemm_wmma<false><<<dim3(25,1,128), 256, 0, stream>>>(
      qh, kh, logit, nullptr, NSP, NKD, 64, 512, 64, NKD,
      8, (long long)NSP*512, 64, 8LL*4096, 4096,
      8LL*NSP*NKD, (long long)NSP*NKD);
  // 8) softmax (+scale, +rel_bias) -> attn f16
  k_softmax<<<(int)((128LL*NSP)/8), 256, 0, stream>>>(logit, rel_bias, attnp);
  // 9) DLA expand (8->24) + GN1 stats + swish
  k_expand<<<dim3(NCHUNK, BATCH*24), 256, 0, stream>>>(attnp, expand_w, a1, psum, psq);
  k_gn_stats<<<BATCH*3, 128, 0, stream>>>(psum, psq, 24, 8, 3, 8.f*(float)SSP, stats);
  k_gn_apply<<<8192, 256, 0, stream>>>(a1, stats, gn1_s, gn1_b, 24, 8, 1, 16LL*24*SSP);
  // 10) depthwise 3x3 + GN2 + swish
  k_dwconv<<<dim3(NCHUNK, BATCH*24), 256, 0, stream>>>(a1, dw_w, a2, psum, psq);
  k_gn_stats<<<BATCH*3, 128, 0, stream>>>(psum, psq, 24, 8, 3, 8.f*(float)SSP, stats);
  k_gn_apply<<<8192, 256, 0, stream>>>(a2, stats, gn2_s, gn2_b, 24, 8, 1, 16LL*24*SSP);
  // 11) channel reduce (24->8) + GN3 + pad to K=64 f16
  k_chreduce<<<dim3(NCHUNK, BATCH*8), 256, 0, stream>>>(a2, reduce_w, a3, psum, psq);
  k_gn_stats<<<BATCH, 128, 0, stream>>>(psum, psq, 8, 8, 1, 8.f*(float)SSP, stats);
  k_gn3_pad<<<(int)((128LL*NSP*64)/256), 256, 0, stream>>>(a3, stats, gn3_s, gn3_b, avh);
  // 12) attn@v: per (b,h): (3136 x 64pad) * v -> scattered into (B*N, C) f16
  k_gemm_wmma<true><<<dim3(25,1,128), 256, 0, stream>>>(
      avh, vh, aoh, nullptr, NSP, 64, 64, 64, 64, 512,
      8, 8LL*NSP*64, (long long)NSP*64, 8LL*4096, 4096,
      (long long)NSP*512, 64);
  // 13) out projection + bias -> d_out f32 (raw (B,N,C) layout == output)
  k_gemm_wmma<false><<<dim3(392,8,1), 256, 0, stream>>>(
      aoh, wpt, out, proj_b, (int)BN, 512, 512, 512, 512, 512,
      1, 0,0, 0,0, 0,0);
}
